// EarthAttention2D_13529146982777
// MI455X (gfx1250) — compile-verified, matrix operand order checked by reading, not run on hardware
//
#include <hip/hip_runtime.h>
#include <hip/hip_bf16.h>

typedef __attribute__((ext_vector_type(16))) __bf16 v16bf;
typedef __attribute__((ext_vector_type(8)))  float  v8f;
typedef __attribute__((ext_vector_type(4)))  __bf16 v4bf;
typedef __attribute__((ext_vector_type(4)))  unsigned int v4u;

union Frag16 { v16bf bf; v4u u[2]; };

__device__ __forceinline__ __bf16 f2bf(float f) {
  unsigned u = __builtin_bit_cast(unsigned, f);
  unsigned r = (u + 0x7FFFu + ((u >> 16) & 1u)) >> 16;   // round-to-nearest-even
  return __builtin_bit_cast(__bf16, (unsigned short)r);
}

// CDNA5 async global->LDS copy (16B). Tracked by ASYNCcnt, bypasses VGPRs.
// LDS address = generic pointer truncated to 32 bits (ISA: LDS addr = addr[31:0],
// i.e. the addrspace(3) offset from LDS_BASE).
__device__ __forceinline__ void async_copy_b128(const void* gptr, void* lptr) {
  unsigned lds = (unsigned)(reinterpret_cast<uintptr_t>(lptr));
  unsigned long long ga = (unsigned long long)(reinterpret_cast<uintptr_t>(gptr));
  asm volatile("global_load_async_to_lds_b128 %0, %1, off"
               :: "v"(lds), "v"(ga) : "memory");
}
__device__ __forceinline__ void wait_async0() {
  asm volatile("s_wait_asynccnt 0" ::: "memory");
}

// Load a 16x32 bf16 WMMA A/B fragment from an LDS tile stored row-major
// [rows][rowStride]. Per CDNA5 ISA layout:
//   lanes 0-15  hold K = {0..7, 16..23}, lanes 16-31 hold K = {8..15, 24..31}
// i.e. 16B chunk at (k0 + half*8) and a second 16B chunk 16 elements later.
__device__ __forceinline__ v16bf lds_frag(const __bf16* tile, int r0, int k0,
                                          int rowStride, int lane) {
  int row = r0 + (lane & 15);
  int half8 = ((lane >> 4) & 1) * 8;
  const __bf16* p = tile + row * rowStride + k0 + half8;
  Frag16 f;
  f.u[0] = *(const v4u*)(p);
  f.u[1] = *(const v4u*)(p + 16);
  return f.bf;
}

__device__ __forceinline__ v8f wmma_bf16(v16bf a, v16bf b, v8f c) {
  return __builtin_amdgcn_wmma_f32_16x16x32_bf16(false, a, false, b,
                                                 (short)0, c, false, false);
}

// ---------------------------------------------------------------- utilities
__global__ __launch_bounds__(256) void cvt_bf16_kernel(const float* __restrict__ src,
                                                       __bf16* __restrict__ dst, int n) {
  int i = blockIdx.x * 256 + threadIdx.x;
  if (i < n) dst[i] = f2bf(src[i]);
}

// bias_eff[head][n][m] = bias_table[pos_index[n*128+m]][TOW/2=4][head]
// bias_table row stride = TOW*HEADS = 128
__global__ __launch_bounds__(256) void bias_gather_kernel(const float* __restrict__ table,
                                                          const int* __restrict__ pidx,
                                                          float* __restrict__ be) {
  int i = blockIdx.x * 256 + threadIdx.x;     // 16*128*128 total
  int head = i >> 14;
  int nm   = i & 16383;
  int idx  = pidx[nm];
  be[head * 16384 + nm] = table[idx * 128 + 4 * 16 + head];
}

// ---------------------------------------------------------------- GEMM
// C[M,N] = A[M,512] @ B[N,512]^T  (+bias).  MODE 0: A fp32 (x), epilogue
// scatters Q (scaled), K, V^T as bf16.  MODE 1: A bf16 (O), fp32 out + bias.
template <int MODE, int NDIM>
__global__ __launch_bounds__(256) void gemm_kernel(const void* __restrict__ Aptr,
                                                   const __bf16* __restrict__ Bw,
                                                   const float* __restrict__ bias,
                                                   __bf16* __restrict__ outQ,
                                                   __bf16* __restrict__ outK,
                                                   __bf16* __restrict__ outV,
                                                   float* __restrict__ outF) {
  constexpr int KD = 512;
  __shared__ __bf16 As[128 * 32];
  __shared__ __bf16 Bs[128 * 32];
  const int t    = threadIdx.x;
  const int lane = t & 31;
  const int w    = t >> 5;
  const int mblk = blockIdx.y * 128;
  const int nblk = blockIdx.x * 128;
  const int wm = (w >> 1) * 32;   // 4 waves along M
  const int wn = (w & 1) * 64;    // 2 waves along N

  v8f acc[2][4];
#pragma unroll
  for (int m = 0; m < 2; ++m)
#pragma unroll
    for (int n = 0; n < 4; ++n) acc[m][n] = (v8f){0.f,0.f,0.f,0.f,0.f,0.f,0.f,0.f};

  for (int kt = 0; kt < KD / 32; ++kt) {
    // --- stage B tile (128x32 bf16 weights) via async direct-to-LDS ---
#pragma unroll
    for (int i = 0; i < 2; ++i) {
      int idx = i * 256 + t;            // 512 16B chunks
      int row = idx >> 2;
      int c8  = (idx & 3) * 8;
      async_copy_b128(Bw + (size_t)(nblk + row) * KD + kt * 32 + c8,
                      Bs + row * 32 + c8);
    }
    // --- stage A tile (128x32) ---
    if (MODE == 0) {
      // fp32 source: must pass through VGPRs for bf16 conversion
      const float* A = (const float*)Aptr;
#pragma unroll
      for (int i = 0; i < 4; ++i) {
        int idx = i * 256 + t;            // 1024 float4 chunks
        int row = idx >> 3;
        int c4  = (idx & 7) * 4;
        float4 v = *(const float4*)(A + (size_t)(mblk + row) * KD + kt * 32 + c4);
        v4bf o;
        o[0] = f2bf(v.x); o[1] = f2bf(v.y); o[2] = f2bf(v.z); o[3] = f2bf(v.w);
        *(v4bf*)(As + row * 32 + c4) = o;
      }
    } else {
      // bf16 source: async direct-to-LDS
      const __bf16* A = (const __bf16*)Aptr;
#pragma unroll
      for (int i = 0; i < 2; ++i) {
        int idx = i * 256 + t;
        int row = idx >> 2;
        int c8  = (idx & 3) * 8;
        async_copy_b128(A + (size_t)(mblk + row) * KD + kt * 32 + c8,
                        As + row * 32 + c8);
      }
    }
    wait_async0();          // barrier does NOT wait ASYNCcnt
    __syncthreads();

    v16bf af[2], bfr[4];
#pragma unroll
    for (int m = 0; m < 2; ++m) af[m] = lds_frag(As, wm + m * 16, 0, 32, lane);
#pragma unroll
    for (int n = 0; n < 4; ++n) bfr[n] = lds_frag(Bs, wn + n * 16, 0, 32, lane);
#pragma unroll
    for (int m = 0; m < 2; ++m)
#pragma unroll
      for (int n = 0; n < 4; ++n) acc[m][n] = wmma_bf16(af[m], bfr[n], acc[m][n]);
    __syncthreads();
  }

  // --- epilogue ---
  const float scale = 0.17677669529663687f;  // 1/sqrt(32)
#pragma unroll
  for (int m = 0; m < 2; ++m) {
#pragma unroll
    for (int n = 0; n < 4; ++n) {
      int c = nblk + wn + n * 16 + (lane & 15);
      float bv = bias[c];
#pragma unroll
      for (int j = 0; j < 8; ++j) {
        int r = mblk + wm + m * 16 + j + ((lane >> 4) & 1) * 8;
        float v = acc[m][n][j] + bv;
        if (MODE == 0) {
          int part = c >> 9;          // 0=q 1=k 2=v
          int head = (c >> 5) & 15;
          int d    = c & 31;
          int win  = r >> 7;
          int nn   = r & 127;
          int wh   = win * 16 + head;
          if (part == 0)      outQ[(size_t)(wh * 128 + nn) * 32 + d] = f2bf(v * scale);
          else if (part == 1) outK[(size_t)(wh * 128 + nn) * 32 + d] = f2bf(v);
          else                outV[(size_t)(wh * 32 + d) * 128 + nn] = f2bf(v);  // V^T
        } else {
          outF[(size_t)r * NDIM + c] = v;
        }
      }
    }
  }
}

// ---------------------------------------------------------------- attention
// One block per (window, head). 8 waves; wave w owns query rows [16w,16w+16).
__global__ __launch_bounds__(256) void attn_kernel(const __bf16* __restrict__ Q,
                                                   const __bf16* __restrict__ Kt,
                                                   const __bf16* __restrict__ Vt,
                                                   const float* __restrict__ be_all,
                                                   __bf16* __restrict__ O) {
  __shared__ __bf16 Qs[128 * 32];
  __shared__ __bf16 Ks[128 * 32];
  __shared__ __bf16 Vts[32 * 128];
  __shared__ __bf16 Ps[8 * 16 * 128];
  const int wh   = blockIdx.x;
  const int head = wh & 15;
  const int win  = wh >> 4;
  const int t    = threadIdx.x;
  const __bf16* Qg = Q  + (size_t)wh * 4096;
  const __bf16* Kg = Kt + (size_t)wh * 4096;
  const __bf16* Vg = Vt + (size_t)wh * 4096;
  // async direct-to-LDS staging of Q, K, V^T tiles (8 KB each)
#pragma unroll
  for (int i = 0; i < 2; ++i) {
    int idx = (i * 256 + t) * 8;   // 512 chunks of 8 bf16
    async_copy_b128(Qg + idx, Qs + idx);
    async_copy_b128(Kg + idx, Ks + idx);
    async_copy_b128(Vg + idx, Vts + idx);
  }
  wait_async0();
  __syncthreads();

  const int lane = t & 31;
  const int w    = t >> 5;
  const int half = (lane >> 4) & 1;
  const int m0   = w * 16;
  const int col  = lane & 15;

  // S = Q @ K^T : K-dim = head_dim = 32 -> exactly one WMMA per 16x16 tile.
  // Preload ALL fragments first so the 8 WMMAs issue back-to-back instead of
  // draining DScnt to 0 before each one.
  v16bf aQ = lds_frag(Qs, m0, 0, 32, lane);
  v16bf bK[8];
#pragma unroll
  for (int nt = 0; nt < 8; ++nt) bK[nt] = lds_frag(Ks, nt * 16, 0, 32, lane);
  v8f s[8];
#pragma unroll
  for (int nt = 0; nt < 8; ++nt) {
    v8f z = (v8f){0.f,0.f,0.f,0.f,0.f,0.f,0.f,0.f};
    s[nt] = wmma_bf16(aQ, bK[nt], z);
  }

  // + relative position bias, then softmax over m (128 cols)
  const float* be = be_all + head * 16384;
  float rmax[8];
#pragma unroll
  for (int j = 0; j < 8; ++j) rmax[j] = -3.0e38f;
#pragma unroll
  for (int nt = 0; nt < 8; ++nt)
#pragma unroll
    for (int j = 0; j < 8; ++j) {
      int nrow = m0 + j + half * 8;
      int mcol = nt * 16 + col;
      float v = s[nt][j] + be[nrow * 128 + mcol];
      s[nt][j] = v;
      rmax[j] = fmaxf(rmax[j], v);
    }
#pragma unroll
  for (int j = 0; j < 8; ++j)
#pragma unroll
    for (int off = 8; off >= 1; off >>= 1)
      rmax[j] = fmaxf(rmax[j], __shfl_xor(rmax[j], off, 32));  // stays in 16-lane half

  float rsum[8];
#pragma unroll
  for (int j = 0; j < 8; ++j) rsum[j] = 0.f;
#pragma unroll
  for (int nt = 0; nt < 8; ++nt)
#pragma unroll
    for (int j = 0; j < 8; ++j) {
      float e = __builtin_amdgcn_exp2f((s[nt][j] - rmax[j]) * 1.4426950408889634f);
      s[nt][j] = e;
      rsum[j] += e;
    }
#pragma unroll
  for (int j = 0; j < 8; ++j)
#pragma unroll
    for (int off = 8; off >= 1; off >>= 1)
      rsum[j] += __shfl_xor(rsum[j], off, 32);
  float rinv[8];
#pragma unroll
  for (int j = 0; j < 8; ++j) rinv[j] = __builtin_amdgcn_rcpf(rsum[j]);

  // stage P (this wave's 16x128 rows) as bf16 in LDS for the PV matmul
  __bf16* Pw = Ps + w * (16 * 128);
#pragma unroll
  for (int nt = 0; nt < 8; ++nt)
#pragma unroll
    for (int j = 0; j < 8; ++j)
      Pw[(j + half * 8) * 128 + nt * 16 + col] = f2bf(s[nt][j] * rinv[j]);
  // same-wave LDS RAW: compiler inserts s_wait_dscnt before the frag reads

  // O = P @ V^T : K = 128 (4 steps), N = 32 (2 tiles).
  // Preload all 12 fragments, then run the 8-WMMA chain.
  v16bf aP[4], bV0[4], bV1[4];
#pragma unroll
  for (int kk = 0; kk < 4; ++kk) {
    aP[kk]  = lds_frag(Pw, 0, kk * 32, 128, lane);
    bV0[kk] = lds_frag(Vts, 0,  kk * 32, 128, lane);
    bV1[kk] = lds_frag(Vts, 16, kk * 32, 128, lane);
  }
  v8f o0 = (v8f){0.f,0.f,0.f,0.f,0.f,0.f,0.f,0.f};
  v8f o1 = (v8f){0.f,0.f,0.f,0.f,0.f,0.f,0.f,0.f};
#pragma unroll
  for (int kk = 0; kk < 4; ++kk) {
    o0 = wmma_bf16(aP[kk], bV0[kk], o0);
    o1 = wmma_bf16(aP[kk], bV1[kk], o1);
  }
  // write O in [b, n, head*32 + d] layout (bf16) for the proj GEMM
#pragma unroll
  for (int j = 0; j < 8; ++j) {
    size_t rb = (size_t)(win * 128 + m0 + j + half * 8) * 512 + head * 32;
    O[rb + col]      = f2bf(o0[j]);
    O[rb + 16 + col] = f2bf(o1[j]);
  }
}

// ---------------------------------------------------------------- launch
extern "C" void kernel_launch(void* const* d_in, const int* in_sizes, int n_in,
                              void* d_out, int out_size, void* d_ws, size_t ws_size,
                              hipStream_t stream) {
  const float* x      = (const float*)d_in[0];
  const float* qkv_w  = (const float*)d_in[1];
  const float* qkv_b  = (const float*)d_in[2];
  const float* proj_w = (const float*)d_in[3];
  const float* proj_b = (const float*)d_in[4];
  const float* btab   = (const float*)d_in[5];
  const int*   pidx   = (const int*)d_in[6];
  (void)in_sizes; (void)n_in; (void)out_size; (void)ws_size;

  char* ws = (char*)d_ws;
  const size_t QKV_ELEMS = (size_t)2048 * 16 * 128 * 32;     // per Q/K/V tensor
  __bf16* qkvw_bf  = (__bf16*)(ws);                           // 1.5 MiB
  __bf16* projw_bf = (__bf16*)(ws + 1572864);                 // 0.5 MiB
  float*  bias_eff = (float*)(ws + 2097152);                  // 1 MiB
  __bf16* Qt = (__bf16*)(ws + 3145728);
  __bf16* Kt = Qt + QKV_ELEMS;
  __bf16* Vt = Kt + QKV_ELEMS;
  __bf16* Ot = Vt + QKV_ELEMS;                                // [B*N, 512] bf16

  cvt_bf16_kernel<<<(1536 * 512) / 256, 256, 0, stream>>>(qkv_w, qkvw_bf, 1536 * 512);
  cvt_bf16_kernel<<<(512 * 512) / 256, 256, 0, stream>>>(proj_w, projw_bf, 512 * 512);
  bias_gather_kernel<<<(16 * 128 * 128) / 256, 256, 0, stream>>>(btab, pidx, bias_eff);

  // QKV: M = 262144, N = 1536
  gemm_kernel<0, 1536><<<dim3(12, 2048), 256, 0, stream>>>(
      x, qkvw_bf, qkv_b, Qt, Kt, Vt, nullptr);

  // attention: one block per (window, head)
  attn_kernel<<<2048 * 16, 256, 0, stream>>>(Qt, Kt, Vt, bias_eff, Ot);

  // proj: M = 262144, N = 512, fp32 out
  gemm_kernel<1, 512><<<dim3(4, 2048), 256, 0, stream>>>(
      Ot, projw_bf, proj_b, nullptr, nullptr, nullptr, (float*)d_out);
}
